// QuantizedConv1d_21663815041807
// MI455X (gfx1250) — matrix operand order repaired and checked by reading, not compiled, not run on hardware
//
#include <hip/hip_runtime.h>
#include <stdint.h>

// CDNA5 / gfx1250 WMMA types (wave32)
typedef __attribute__((ext_vector_type(16))) __bf16 v16bf;
typedef __attribute__((ext_vector_type(8)))  __bf16 v8bf16;
typedef __attribute__((ext_vector_type(4)))  __bf16 v4bf16;
typedef __attribute__((ext_vector_type(8)))  float  v8f;
typedef __attribute__((ext_vector_type(4)))  int    v4i;

#define BATCH 16
#define CIN   1024
#define COUT  1024
#define LEN   4096

#define MT 128            // M tile (C_OUT)
#define NT 128            // N tile (L)
#define KT 64             // K tile (C_IN): two 32-deep WMMA steps
#define KTS (KT + 8)      // sXT row stride (bf16): 144 B, 16B-aligned, staggers banks

#if __has_builtin(__builtin_amdgcn_global_load_async_to_lds_b128)
#define HAVE_ASYNC_LDS 1
#endif

// ---------------- prep: dequant W int8 -> raw-integer bf16 (exact, scale applied later)
__global__ __launch_bounds__(256) void dequant_w_bf16(
    const int8_t* __restrict__ qw, __bf16* __restrict__ wbf)
{
    int i = (blockIdx.x * 256 + threadIdx.x) * 4;
    int32_t pkt = *(const int32_t*)(qw + i);
    const int8_t* pb = (const int8_t*)&pkt;
    v4bf16 o;
#pragma unroll
    for (int j = 0; j < 4; ++j)
        o[j] = (__bf16)(float)pb[j];
    *(v4bf16*)(wbf + i) = o;
}

// ---------------- main GEMM: out[b] = scale * (W_int . x_bf16) + bias
__global__ __launch_bounds__(256) void qconv1d_wmma_bf16(
    const float*  __restrict__ x,        // [B, CIN, LEN] fp32
    const __bf16* __restrict__ wbf,      // [COUT, CIN] bf16 (raw int values)
    const int8_t* __restrict__ qb,       // [COUT] int8
    const float*  __restrict__ scale_p,
    const float*  __restrict__ bscale_p,
    float*        __restrict__ out)      // [B, COUT, LEN] fp32
{
    __shared__ __bf16 sA [MT * KT];      // [m][k], k contiguous          (16 KB)
    __shared__ __bf16 sXT[NT * KTS];     // [n][k], k contiguous, bf16    (18 KB)

    const int t      = threadIdx.x;
    const int lane   = t & 31;
    const int wave   = t >> 5;
    const int laneLo = lane & 15;
    const int half   = lane >> 4;        // wave32 half

    const int waveM  = wave & 3;         // 4 M-groups of 32 rows
    const int waveN  = wave >> 2;        // 2 N-groups of 64 cols
    const int mBase  = waveM * 32;
    const int nBase  = waveN * 64;

    const int mBlock = blockIdx.x * MT;  // M fastest -> x panel reused from L2
    const int lBlock = blockIdx.y * NT;
    const int b      = blockIdx.z;

    const size_t xBase = (size_t)b * CIN * LEN;

    v8f acc[2][4];
    const v8f vzero = {0.f, 0.f, 0.f, 0.f, 0.f, 0.f, 0.f, 0.f};
#pragma unroll
    for (int mi = 0; mi < 2; ++mi)
#pragma unroll
        for (int ni = 0; ni < 4; ++ni)
            acc[mi][ni] = vzero;

    // x staging assignment: thread owns one column l and 32 k-values
    const int xcol   = t & 127;          // 0..127  (lane-consecutive -> coalesced)
    const int kcBase = (t >> 7) * 32;    // 0 or 32

    for (int k0 = 0; k0 < CIN; k0 += KT) {
        __syncthreads();   // previous tile fully consumed

        // ---- A tile: raw 16B copies of bf16 weights (no VALU); async if available
#if defined(HAVE_ASYNC_LDS)
#pragma unroll
        for (int c = 0; c < 4; ++c) {
            int cid = t + c * 256;                 // 1024 chunks of 16 B
            int row = cid >> 3;
            int col = (cid & 7) * 8;               // bf16 units
            const __bf16* gp = wbf + (size_t)(mBlock + row) * CIN + k0 + col;
            __bf16*       lp = &sA[row * KT + col];
            __builtin_amdgcn_global_load_async_to_lds_b128(
                (__attribute__((address_space(1))) v4i*)gp,
                (__attribute__((address_space(3))) v4i*)lp,
                0, 0);
        }
#else
#pragma unroll
        for (int c = 0; c < 4; ++c) {
            int cid = t + c * 256;
            int row = cid >> 3;
            int col = (cid & 7) * 8;
            int4 pkt = *(const int4*)(wbf + (size_t)(mBlock + row) * CIN + k0 + col);
            *(int4*)&sA[row * KT + col] = pkt;
        }
#endif

        // ---- x tile: coalesced column loads, convert fp32->bf16 once,
        //      store transposed [n][k] so B fragments are contiguous b128 runs
        {
            v8bf16 pk[4];
#pragma unroll
            for (int c = 0; c < 32; ++c) {
                float f = x[xBase + (size_t)(k0 + kcBase + c) * LEN + lBlock + xcol];
                pk[c >> 3][c & 7] = (__bf16)f;
            }
#pragma unroll
            for (int c = 0; c < 4; ++c)
                *(v8bf16*)&sXT[xcol * KTS + kcBase + c * 8] = pk[c];
        }

#if defined(HAVE_ASYNC_LDS)
#if __has_builtin(__builtin_amdgcn_s_wait_asynccnt)
        __builtin_amdgcn_s_wait_asynccnt(0);
#else
        asm volatile("s_wait_asynccnt 0x0" ::: "memory");
#endif
#endif
        __syncthreads();

        // ---- two 32-deep WMMA steps over this K tile
#pragma unroll
        for (int kk = 0; kk < KT; kk += 32) {
            v16bf afrag[2];
            v16bf bfrag[4];

            // A fragment (16x32 bf16): per lane row = laneLo,
            // elements: K = kk + 8*half + e (e<8), K = kk + 16 + 8*half + (e-8)
#pragma unroll
            for (int mi = 0; mi < 2; ++mi) {
                const __bf16* ap = &sA[(mBase + mi * 16 + laneLo) * KT + kk + half * 8];
                v8bf16 lo = *(const v8bf16*)(ap);
                v8bf16 hi = *(const v8bf16*)(ap + 16);
                afrag[mi] = __builtin_shufflevector(lo, hi,
                    0,1,2,3,4,5,6,7, 8,9,10,11,12,13,14,15);
            }

            // B fragment (32x16 bf16): per lane col = laneLo,
            // K = kk + 8*half + e (e<8), K = kk + 16 + 8*half + (e-8) -> two b128 loads
#pragma unroll
            for (int ni = 0; ni < 4; ++ni) {
                const __bf16* bp =
                    &sXT[(nBase + ni * 16 + laneLo) * KTS + kk + half * 8];
                v8bf16 lo = *(const v8bf16*)(bp);
                v8bf16 hi = *(const v8bf16*)(bp + 16);
                bfrag[ni] = __builtin_shufflevector(lo, hi,
                    0,1,2,3,4,5,6,7, 8,9,10,11,12,13,14,15);
            }

            // 8 WMMAs, full EXEC, f32 accumulate
#pragma unroll
            for (int mi = 0; mi < 2; ++mi)
#pragma unroll
                for (int ni = 0; ni < 4; ++ni)
                    acc[mi][ni] = __builtin_amdgcn_wmma_f32_16x16x32_bf16(
                        false, afrag[mi], false, bfrag[ni],
                        (short)0, acc[mi][ni], false, false);
        }
    }

    // ---- epilogue: weight scale + dequant bias
    const float s  = *scale_p;
    const float bs = *bscale_p;

#pragma unroll
    for (int mi = 0; mi < 2; ++mi) {
#pragma unroll
        for (int r = 0; r < 8; ++r) {
            int o = mBlock + mBase + mi * 16 + r + half * 8;   // D: row = r + 8*half
            float bias = (float)qb[o] * bs;
#pragma unroll
            for (int ni = 0; ni < 4; ++ni) {
                int l = lBlock + nBase + ni * 16 + laneLo;     // D: col = laneLo
                out[((size_t)b * COUT + o) * LEN + l] = acc[mi][ni][r] * s + bias;
            }
        }
    }
}

extern "C" void kernel_launch(void* const* d_in, const int* in_sizes, int n_in,
                              void* d_out, int out_size, void* d_ws, size_t ws_size,
                              hipStream_t stream) {
    (void)in_sizes; (void)n_in; (void)out_size; (void)ws_size;
    const float*  x   = (const float*)d_in[0];
    const int8_t* qw  = (const int8_t*)d_in[1];
    const int8_t* qb  = (const int8_t*)d_in[2];
    const float*  sc  = (const float*)d_in[3];
    const float*  bsc = (const float*)d_in[4];
    float*  out = (float*)d_out;
    __bf16* wbf = (__bf16*)d_ws;         // 2 MB scratch for bf16 weights

    // 1) dequant weights once (1M int8 -> bf16), stream-ordered before GEMM
    hipLaunchKernelGGL(dequant_w_bf16, dim3((COUT * CIN) / (256 * 4)), dim3(256),
                       0, stream, qw, wbf);

    // 2) tiled WMMA GEMM
    dim3 grid(COUT / MT, LEN / NT, BATCH);
    dim3 block(256);
    hipLaunchKernelGGL(qconv1d_wmma_bf16, grid, block, 0, stream,
                       x, wbf, qb, sc, bsc, out);
}
